// MyRnn_30099130810553
// MI455X (gfx1250) — matrix operand order repaired and checked by reading, not compiled, not run on hardware
//
#include <hip/hip_runtime.h>

// ---------------------------------------------------------------------------
// SimpleRNN inference for MI455X (gfx1250), wave32 + WMMA bf16 split-precision.
//   x  = emb[ids]                               [B,T,100]
//   XW = x @ W_xh + b_h        (fp32 in ws)     [T,B,512]  (T-major for scan)
//   h_{t+1} = tanh(XW_t + h_t @ W_hh)           80 sequential steps
//   out = sigmoid(h_79 @ W_out + b_out)         [B]
// GEMMs run as split-bf16 (hi+lo limbs, 3x v_wmma_f32_16x16x32_bf16 per K=32).
// k_step stages W_hh^T tiles in LDS via GLOBAL_LOAD_ASYNC_TO_LDS_B128 with
// double buffering (ASYNCcnt-tracked), overlapping the fetch with WMMAs.
// ---------------------------------------------------------------------------

typedef __bf16 bf16_t;
typedef __attribute__((ext_vector_type(16))) __bf16 v16bf;
typedef __attribute__((ext_vector_type(8)))  __bf16 v8bf;
typedef __attribute__((ext_vector_type(8)))  float  v8f;
typedef __attribute__((ext_vector_type(4)))  float  v4f;

#define WMMA_BF16(a, b, c) \
  __builtin_amdgcn_wmma_f32_16x16x32_bf16(false, (a), false, (b), (short)0, (c), false, false)

constexpr int BATCH = 2048;
constexpr int TLEN  = 80;
constexpr int EMB   = 100;
constexpr int UNITS = 512;
constexpr int KPAD  = 128;                 // EMB padded to multiple of 32
constexpr int MTOK  = BATCH * TLEN;        // 163840 token rows

// ---------------------------------------------------------------------------
// Prep: build N-major, hi/lo-split bf16 weight tables.
//   WxT[n*KPAD + k] = (k < 100) ? W_xh[k*512 + n] : 0
//   WhT[n*512  + k] =             W_hh[k*512 + n]
// ---------------------------------------------------------------------------
__global__ void k_prep(const float* __restrict__ W_xh, const float* __restrict__ W_hh,
                       bf16_t* __restrict__ WxT_hi, bf16_t* __restrict__ WxT_lo,
                       bf16_t* __restrict__ WhT_hi, bf16_t* __restrict__ WhT_lo) {
  int idx = blockIdx.x * blockDim.x + threadIdx.x;
  if (idx < UNITS * UNITS) {
    int n = idx >> 9, k = idx & 511;
    float v = W_hh[k * UNITS + n];
    bf16_t hi = (bf16_t)v;
    WhT_hi[idx] = hi;
    WhT_lo[idx] = (bf16_t)(v - (float)hi);
  }
  if (idx < UNITS * KPAD) {
    int n = idx >> 7, k = idx & 127;
    float v = (k < EMB) ? W_xh[k * UNITS + n] : 0.0f;
    bf16_t hi = (bf16_t)v;
    WxT_hi[idx] = hi;
    WxT_lo[idx] = (bf16_t)(v - (float)hi);
  }
}

// Load 8 consecutive fp32 from a 100-wide embedding row with zero padding.
__device__ inline void load8_pad(const float* __restrict__ row, int k0, float* out8) {
  if (k0 + 8 <= EMB) {
    v4f a = *(const v4f*)(row + k0);
    v4f b = *(const v4f*)(row + k0 + 4);
#pragma unroll
    for (int i = 0; i < 4; ++i) { out8[i] = a[i]; out8[4 + i] = b[i]; }
  } else {
#pragma unroll
    for (int i = 0; i < 8; ++i) { int k = k0 + i; out8[i] = (k < EMB) ? row[k] : 0.0f; }
  }
}

// ---------------------------------------------------------------------------
// Input projection: XW[r, u] = emb[ids[r]] @ W_xh + b_h, r = t*B + b.
// Block = 128 threads (4 waves). Wave -> 16(M) x 64(N), WG -> 64 x 64.
// ---------------------------------------------------------------------------
__global__ void k_proj(const int* __restrict__ ids, const float* __restrict__ emb,
                       const bf16_t* __restrict__ WxT_hi, const bf16_t* __restrict__ WxT_lo,
                       const float* __restrict__ b_h, float* __restrict__ XW) {
  const int lane = threadIdx.x & 31;
  const int wave = threadIdx.x >> 5;
  const int tileM = blockIdx.x * 64 + wave * 16;
  const int tileN = blockIdx.y * 64;
  const int lidx  = lane & 15;
  const int half  = lane >> 4;          // 0: K lo-half of chunk, 1: K hi-half

  // token row handled by this lane's A-fragment rows
  const int mrow = tileM + lidx;
  const int t    = mrow / BATCH;
  const int b    = mrow - t * BATCH;
  const int tok  = ids[b * TLEN + t];
  const float* erow = emb + (size_t)tok * EMB;

  // pull the whole 400B embedding row toward the caches; reused by all 4 ktiles
  __builtin_prefetch(erow, 0, 3);
  __builtin_prefetch(erow + 64, 0, 3);

  v8f c[4];
#pragma unroll
  for (int j = 0; j < 4; ++j) c[j] = (v8f){0.f, 0.f, 0.f, 0.f, 0.f, 0.f, 0.f, 0.f};

#pragma unroll
  for (int kt = 0; kt < KPAD / 32; ++kt) {
    // ---- A fragment (16x32 bf16, split hi/lo), gathered from fp32 embeddings
    const int kb = kt * 32 + half * 16;
    float vals[16];
    load8_pad(erow, kb, vals);
    load8_pad(erow, kb + 8, vals + 8);
    v16bf a_hi, a_lo;
#pragma unroll
    for (int i = 0; i < 16; ++i) {
      bf16_t hi = (bf16_t)vals[i];
      a_hi[i] = hi;
      a_lo[i] = (bf16_t)(vals[i] - (float)hi);
    }

    // ---- 4 N-tiles, B fragments contiguous per lane from N-major table
#pragma unroll
    for (int j = 0; j < 4; ++j) {
      const int ncol = tileN + j * 16 + lidx;
      const size_t boff = (size_t)ncol * KPAD + kt * 32 + half * 16;
      v16bf b_hi = *(const v16bf*)(WxT_hi + boff);
      v16bf b_lo = *(const v16bf*)(WxT_lo + boff);
      c[j] = WMMA_BF16(a_hi, b_hi, c[j]);
      c[j] = WMMA_BF16(a_hi, b_lo, c[j]);
      c[j] = WMMA_BF16(a_lo, b_hi, c[j]);
    }
  }

  // ---- epilogue: + b_h, fp32 store. C layout: vgpr i -> M = i + 8*half, N = lane&15
#pragma unroll
  for (int j = 0; j < 4; ++j) {
    const int col = tileN + j * 16 + lidx;
    const float bias = b_h[col];
#pragma unroll
    for (int i = 0; i < 8; ++i) {
      const int row = tileM + i + half * 8;
      XW[(size_t)row * UNITS + col] = c[j][i] + bias;
    }
  }
}

// Issue 64B (4 x b128) global -> LDS async copy for this thread's column.
// Per ISA: the offset is added to both the LDS and global address; each load counts on ASYNCcnt.
__device__ inline void async_fill64(const bf16_t* __restrict__ src, bf16_t* dst) {
  unsigned           l = (unsigned)(uintptr_t)dst;          // flat shared -> LDS offset
  unsigned long long g = (unsigned long long)(uintptr_t)src;
  asm volatile(
      "global_load_async_to_lds_b128 %0, %1, off\n\t"
      "global_load_async_to_lds_b128 %0, %1, off offset:16\n\t"
      "global_load_async_to_lds_b128 %0, %1, off offset:32\n\t"
      "global_load_async_to_lds_b128 %0, %1, off offset:48"
      :: "v"(l), "v"(g) : "memory");
}

// ---------------------------------------------------------------------------
// One recurrence step: h_out = tanh(XW_t + h_in @ W_hh).
// Block = 128 threads (4 waves), WG tile 64(M) x 64(N).
// W_hh^T tile double-buffered in LDS via async global->LDS loads.
// ---------------------------------------------------------------------------
__global__ void k_step(const bf16_t* __restrict__ h_hi, const bf16_t* __restrict__ h_lo,
                       const bf16_t* __restrict__ WhT_hi, const bf16_t* __restrict__ WhT_lo,
                       const float* __restrict__ XWt,
                       bf16_t* __restrict__ ho_hi, bf16_t* __restrict__ ho_lo) {
  __shared__ alignas(32) bf16_t Bs[2][2][64 * 32];  // [buf][hi|lo][n_local*32 + k_local]

  const int tid  = threadIdx.x;
  const int lane = tid & 31;
  const int wave = tid >> 5;
  const int tileM = blockIdx.x * 64 + wave * 16;
  const int tileN = blockIdx.y * 64;
  const int lidx  = lane & 15;
  const int half  = lane >> 4;
  constexpr int KT = UNITS / 32;          // 16 ktiles

  // this thread's column of the cooperative fill
  const int fcol   = tid & 63;
  const int fwhich = tid >> 6;            // 0 -> hi limb, 1 -> lo limb
  const bf16_t* ftab = fwhich ? WhT_lo : WhT_hi;
  const bf16_t* fsrc = ftab + (size_t)(tileN + fcol) * UNITS;   // + kt*32 per tile

  v8f c[4];
#pragma unroll
  for (int j = 0; j < 4; ++j) c[j] = (v8f){0.f, 0.f, 0.f, 0.f, 0.f, 0.f, 0.f, 0.f};

  const int mrow = tileM + lidx;
  const bf16_t* arow_hi = h_hi + (size_t)mrow * UNITS;
  const bf16_t* arow_lo = h_lo + (size_t)mrow * UNITS;

  // prologue: async fill of ktile 0 into buffer 0 (4 outstanding per wave)
  async_fill64(fsrc, &Bs[0][fwhich][fcol * 32]);

  for (int kt = 0; kt < KT; ++kt) {
    // issue next tile's fill into the other buffer, then wait for current tile.
    if (kt + 1 < KT) {
      async_fill64(fsrc + (kt + 1) * 32, &Bs[(kt + 1) & 1][fwhich][fcol * 32]);
      asm volatile("s_wait_asynccnt 0x4" ::: "memory");   // current tile done, next in flight
    } else {
      asm volatile("s_wait_asynccnt 0x0" ::: "memory");
    }
    __syncthreads();

    // ---- A fragments (h tile, already split bf16 in memory)
    const int kb = kt * 32 + half * 16;
    v8bf ah0 = *(const v8bf*)(arow_hi + kb);
    v8bf ah1 = *(const v8bf*)(arow_hi + kb + 8);
    v8bf al0 = *(const v8bf*)(arow_lo + kb);
    v8bf al1 = *(const v8bf*)(arow_lo + kb + 8);
    v16bf a_hi = __builtin_shufflevector(ah0, ah1, 0,1,2,3,4,5,6,7,8,9,10,11,12,13,14,15);
    v16bf a_lo = __builtin_shufflevector(al0, al1, 0,1,2,3,4,5,6,7,8,9,10,11,12,13,14,15);

    // ---- 4 N-tiles from LDS
    const bf16_t* bs_hi = Bs[kt & 1][0];
    const bf16_t* bs_lo = Bs[kt & 1][1];
#pragma unroll
    for (int j = 0; j < 4; ++j) {
      const int boff = (j * 16 + lidx) * 32 + half * 16;
      v16bf b_hi = *(const v16bf*)(bs_hi + boff);
      v16bf b_lo = *(const v16bf*)(bs_lo + boff);
      c[j] = WMMA_BF16(a_hi, b_hi, c[j]);
      c[j] = WMMA_BF16(a_hi, b_lo, c[j]);
      c[j] = WMMA_BF16(a_lo, b_hi, c[j]);
    }
    __syncthreads();   // all waves done reading this buffer before it is refilled
  }

  // ---- epilogue: + XW_t, tanh, write split-bf16 h
#pragma unroll
  for (int j = 0; j < 4; ++j) {
    const int col = tileN + j * 16 + lidx;
#pragma unroll
    for (int i = 0; i < 8; ++i) {
      const int row = tileM + i + half * 8;
      float pre = c[j][i] + XWt[(size_t)row * UNITS + col];
      float hv  = tanhf(pre);
      bf16_t hi = (bf16_t)hv;
      ho_hi[(size_t)row * UNITS + col] = hi;
      ho_lo[(size_t)row * UNITS + col] = (bf16_t)(hv - (float)hi);
    }
  }
}

// ---------------------------------------------------------------------------
// Output head: sigmoid(h @ W_out + b_out). One wave per batch row.
// ---------------------------------------------------------------------------
__global__ void k_out(const bf16_t* __restrict__ h_hi, const bf16_t* __restrict__ h_lo,
                      const float* __restrict__ W_out, const float* __restrict__ b_out,
                      float* __restrict__ out) {
  const int lane = threadIdx.x & 31;
  const int wave = threadIdx.x >> 5;
  const int row  = blockIdx.x * 8 + wave;

  float s = 0.0f;
#pragma unroll
  for (int k = lane; k < UNITS; k += 32) {
    float hv = (float)h_hi[(size_t)row * UNITS + k] + (float)h_lo[(size_t)row * UNITS + k];
    s += hv * W_out[k];
  }
#pragma unroll
  for (int off = 16; off > 0; off >>= 1) s += __shfl_down(s, off, 32);
  if (lane == 0) {
    float z = s + b_out[0];
    out[row] = 1.0f / (1.0f + expf(-z));
  }
}

// ---------------------------------------------------------------------------
extern "C" void kernel_launch(void* const* d_in, const int* in_sizes, int n_in,
                              void* d_out, int out_size, void* d_ws, size_t ws_size,
                              hipStream_t stream) {
  const int*   ids   = (const int*)d_in[0];
  const float* emb   = (const float*)d_in[1];
  const float* W_xh  = (const float*)d_in[2];
  const float* W_hh  = (const float*)d_in[3];
  const float* b_h   = (const float*)d_in[4];
  const float* W_out = (const float*)d_in[5];
  const float* b_out = (const float*)d_in[6];
  float* out = (float*)d_out;

  // workspace layout (bytes); total ~345 MB
  char* ws = (char*)d_ws;
  const size_t XW_BYTES  = (size_t)MTOK * UNITS * sizeof(float);        // 335 MB
  const size_t WHT_BYTES = (size_t)UNITS * UNITS * sizeof(bf16_t);      // 512 KB
  const size_t WXT_BYTES = (size_t)UNITS * KPAD  * sizeof(bf16_t);      // 128 KB
  const size_t H_BYTES   = (size_t)BATCH * UNITS * sizeof(bf16_t);      // 2 MB

  float*  XW      = (float*)ws;                       size_t off = XW_BYTES;
  bf16_t* WhT_hi  = (bf16_t*)(ws + off);              off += WHT_BYTES;
  bf16_t* WhT_lo  = (bf16_t*)(ws + off);              off += WHT_BYTES;
  bf16_t* WxT_hi  = (bf16_t*)(ws + off);              off += WXT_BYTES;
  bf16_t* WxT_lo  = (bf16_t*)(ws + off);              off += WXT_BYTES;
  bf16_t* hA_hi   = (bf16_t*)(ws + off);              off += H_BYTES;
  bf16_t* hA_lo   = (bf16_t*)(ws + off);              off += H_BYTES;
  bf16_t* hB_hi   = (bf16_t*)(ws + off);              off += H_BYTES;
  bf16_t* hB_lo   = (bf16_t*)(ws + off);              off += H_BYTES;

  // 1) weight transpose + hi/lo split
  k_prep<<<(UNITS * UNITS + 255) / 256, 256, 0, stream>>>(W_xh, W_hh,
                                                          WxT_hi, WxT_lo, WhT_hi, WhT_lo);
  // 2) h0 = 0 (hA_hi and hA_lo are contiguous)
  hipMemsetAsync(hA_hi, 0, 2 * H_BYTES, stream);

  // 3) input projection into XW [T, B, U] (fp32)
  k_proj<<<dim3(MTOK / 64, UNITS / 64), 128, 0, stream>>>(ids, emb, WxT_hi, WxT_lo, b_h, XW);

  // 4) 80 recurrence steps, ping-pong h buffers (t even: A->B, t odd: B->A)
  for (int t = 0; t < TLEN; ++t) {
    const float* XWt = XW + (size_t)t * BATCH * UNITS;
    if ((t & 1) == 0)
      k_step<<<dim3(BATCH / 64, UNITS / 64), 128, 0, stream>>>(hA_hi, hA_lo, WhT_hi, WhT_lo,
                                                               XWt, hB_hi, hB_lo);
    else
      k_step<<<dim3(BATCH / 64, UNITS / 64), 128, 0, stream>>>(hB_hi, hB_lo, WhT_hi, WhT_lo,
                                                               XWt, hA_hi, hA_lo);
  }
  // TLEN = 80 (even) -> final h is in hA

  // 5) output head
  k_out<<<BATCH / 8, 256, 0, stream>>>(hA_hi, hA_lo, W_out, b_out, out);
}
